// CardClassifier_6914897347019
// MI455X (gfx1250) — matrix operand order repaired and checked by reading, not compiled, not run on hardware
//
#include <hip/hip_runtime.h>
#include <hip/hip_bf16.h>
#include <math.h>

typedef __attribute__((ext_vector_type(16))) _Float16 v16h;
typedef __attribute__((ext_vector_type(8)))  float    v8f;
typedef __attribute__((ext_vector_type(8)))  _Float16 v8h;                    // 16B aligned
typedef _Float16 v8h_u __attribute__((ext_vector_type(8), aligned(2)));       // unaligned-safe
typedef __attribute__((ext_vector_type(4))) unsigned int u32x4;
typedef __attribute__((ext_vector_type(8))) int i32x8;
typedef __attribute__((ext_vector_type(4))) int i32x4;

#define DEV static __device__ __forceinline__

// ==================================================================
// Tensor Data Mover: 2-D tile load Global -> LDS (D# per ISA 8.3/8.4)
//   tile: tile_d1 rows x tile_d0 elements (f16), row stride = stride0 elems
//   rows/cols beyond tensor_d1/tensor_d0 read as zero (free zero-pad).
// 6-arg builtin form (this toolchain): (u32x4, i32x8, i32x4, i32x4, i32x8, i32)
// ==================================================================
DEV void tdm_load_2d_f16(unsigned lds_off, const void* gaddr,
                         unsigned tensor_d0, unsigned tensor_d1,
                         unsigned tile_d0, unsigned tile_d1,
                         unsigned long long stride0_elems) {
  unsigned long long ga = (unsigned long long)gaddr;
  u32x4 g0;
  g0[0] = 1u;                                          // count=1, user mode
  g0[1] = lds_off;                                     // lds_addr
  g0[2] = (unsigned)(ga & 0xffffffffu);                // global_addr[31:0]
  g0[3] = (unsigned)((ga >> 32) & 0x1ffffffu) | (2u << 30);  // [56:32] | type=2
  i32x8 g1;
  g1[0] = (1 << 16);                                   // data_size=1 -> 2 bytes
  g1[1] = (int)((tensor_d0 & 0xffffu) << 16);          // tensor_dim0 lo
  g1[2] = (int)((tensor_d0 >> 16) & 0xffffu) | (int)((tensor_d1 & 0xffffu) << 16);
  g1[3] = (int)((tensor_d1 >> 16) & 0xffffu) | (int)((tile_d0 & 0xffffu) << 16);
  g1[4] = (int)(tile_d1 & 0xffffu);                    // tile_dim1 (tile_dim2=0)
  g1[5] = (int)(unsigned)(stride0_elems & 0xffffffffull);        // dim0_stride lo
  g1[6] = (int)(unsigned)((stride0_elems >> 32) & 0xffffull);    // dim0_stride hi
  g1[7] = 0;
  i32x4 z4 = {0, 0, 0, 0};
  i32x8 z8 = {0, 0, 0, 0, 0, 0, 0, 0};
  __builtin_amdgcn_tensor_load_to_lds(g0, g1, z4, z4, z8, 0);
}

// ------------------------------------------------------------------
// layout converters
// ------------------------------------------------------------------
__global__ void cvt_x_nhwc_kernel(const float* __restrict__ src,  // [B,3,H,W]
                                  _Float16* __restrict__ dst,     // [B,H,W,3]
                                  int Cc, int H, int W, int n) {
  int i = blockIdx.x * blockDim.x + threadIdx.x;
  if (i >= n) return;
  int c = i % Cc;  int t = i / Cc;
  int w = t % W;   t /= W;
  int h = t % H;   int b = t / H;
  dst[i] = (_Float16)src[(((size_t)b * Cc + c) * H + h) * W + w];
}

__global__ void cvt_convw_kernel(const float* __restrict__ src,  // [Cout,Cin,3,3]
                                 _Float16* __restrict__ dst,     // [Cout,9,Cin]
                                 int Cin, int n) {
  int i = blockIdx.x * blockDim.x + threadIdx.x;
  if (i >= n) return;
  int cin = i % Cin;  int t = i / Cin;
  int pos = t % 9;    int cout = t / 9;
  dst[i] = (_Float16)src[((size_t)cout * Cin + cin) * 9 + pos];
}

__global__ void cvt_wT_kernel(const float* __restrict__ src,  // [E,K,N]
                              _Float16* __restrict__ dst,     // [E,N,K]
                              int K, int N, int n) {
  int i = blockIdx.x * blockDim.x + threadIdx.x;
  if (i >= n) return;
  int k = i % K;  int t = i / K;
  int c = t % N;  int e = t / N;
  dst[i] = (_Float16)src[((size_t)e * K + k) * N + c];
}

__global__ void feats_transpose_kernel(const _Float16* __restrict__ src, // [B,F,E] (NHWC)
                                       _Float16* __restrict__ dst,       // [B,E,F]
                                       int E, int F, int n) {
  int i = blockIdx.x * blockDim.x + threadIdx.x;
  if (i >= n) return;
  int f = i % F;  int t = i / F;
  int e = t % E;  int b = t / E;
  dst[i] = src[((size_t)b * F + f) * E + e];
}

// ==================================================================
// Conv3x3(SAME)+bias+relu+2x2maxpool, NHWC, implicit-GEMM WMMA.
// One wave: 16-pixel (4x4 patch) x 16-outchannel tile; pool in regs.
// Fast path (Cin % 32 == 0): fully unrolled 9 taps, vector loads.
// grid: x=(H/4)*(W/4), y=Cout/16, z=B.  32 threads.
// ==================================================================
__global__ void conv_wmma_kernel(const _Float16* __restrict__ in,   // [B,H,W,Cin]
                                 const _Float16* __restrict__ wgt,  // [Cout,9,Cin]
                                 const float* __restrict__ bias,    // [Cout]
                                 _Float16* __restrict__ out,        // [B,H/2,W/2,Cout]
                                 int Cin, int H, int W, int Cout) {
  const int lane = threadIdx.x;
  const int tiles_x = W >> 2;
  const int ty = blockIdx.x / tiles_x, tx = blockIdx.x - ty * tiles_x;
  const int cout0 = blockIdx.y << 4;
  const int b = blockIdx.z;
  const int K = Cin * 9;
  const int l15 = lane & 15;
  const int grp = lane >> 4;
  const int oh = (ty << 2) + (l15 >> 2);
  const int ow = (tx << 2) + (l15 & 3);
  const _Float16* wrow = wgt + (size_t)(cout0 + l15) * K;   // [9][Cin] for this cout

  v8f acc = {};
  if ((Cin & 31) == 0) {
    // -------- fast path: K-step = 32 input channels of one tap --------
#pragma unroll
    for (int pos = 0; pos < 9; ++pos) {
      const int kh = pos / 3, kw = pos % 3;
      const int ih = oh + kh - 1, iw = ow + kw - 1;
      const bool inb = (ih >= 0) & (ih < H) & (iw >= 0) & (iw < W);
      const _Float16* px = in + ((size_t)(b * H + ih) * W + iw) * Cin;
      const _Float16* wp = wrow + pos * Cin;
      for (int c0 = 0; c0 < Cin; c0 += 32) {
        v16h a = {};
        if (inb) {
          v8h lo = *(const v8h*)(px + c0 + (grp << 3));
          v8h hi = *(const v8h*)(px + c0 + 16 + (grp << 3));
#pragma unroll
          for (int i = 0; i < 8; ++i) { a[i] = lo[i]; a[8 + i] = hi[i]; }
        }
        v16h bf;
        {
          v8h lo = *(const v8h*)(wp + c0 + (grp << 3));
          v8h hi = *(const v8h*)(wp + c0 + 16 + (grp << 3));
#pragma unroll
          for (int i = 0; i < 8; ++i) { bf[i] = lo[i]; bf[8 + i] = hi[i]; }
        }
        if (c0 + 32 < Cin)
          __builtin_prefetch((const void*)(wp + c0 + 32), 0, 0);
        acc = __builtin_amdgcn_wmma_f32_16x16x32_f16(false, a, false, bf,
                                                     (short)0, acc, false, false);
      }
    }
  } else {
    // -------- generic path (conv1, Cin=3, K=27): per-element gather --------
    for (int kb = 0; kb < K; kb += 32) {
      v16h a;
#pragma unroll
      for (int v = 0; v < 8; ++v)
#pragma unroll
        for (int p = 0; p < 2; ++p) {
          int k = kb + ((v >> 2) << 4) + (grp << 3) + ((v & 3) << 1) + p;
          _Float16 val = (_Float16)0.f;
          if (k < K) {
            int pos = k / Cin, cin = k - pos * Cin;
            int ih = oh + pos / 3 - 1, iw = ow + pos % 3 - 1;
            if (ih >= 0 && ih < H && iw >= 0 && iw < W)
              val = in[((size_t)(b * H + ih) * W + iw) * Cin + cin];
          }
          a[2 * v + p] = val;
        }
      v16h bf;
#pragma unroll
      for (int j = 0; j < 8; ++j)
#pragma unroll
        for (int p = 0; p < 2; ++p) {
          int k = kb + ((j >> 2) << 4) + (grp << 3) + ((j & 3) << 1) + p;
          bf[2 * j + p] = (k < K) ? wrow[k] : (_Float16)0.f;
        }
      acc = __builtin_amdgcn_wmma_f32_16x16x32_f16(false, a, false, bf,
                                                   (short)0, acc, false, false);
    }
  }
  // ---- epilogue: 2x2 maxpool in registers + bias + relu, NHWC store ----
  float bv = bias[cout0 + l15];
  float p0 = fmaxf(fmaxf(acc[0], acc[1]), fmaxf(acc[4], acc[5])) + bv;
  float p1 = fmaxf(fmaxf(acc[2], acc[3]), fmaxf(acc[6], acc[7])) + bv;
  p0 = fmaxf(p0, 0.f);
  p1 = fmaxf(p1, 0.f);
  const int PH = H >> 1, PW = W >> 1;
  const int ph = (ty << 1) + grp, pw0 = tx << 1;
  size_t o0 = ((size_t)(b * PH + ph) * PW + pw0) * Cout + cout0 + l15;
  out[o0] = (_Float16)p0;
  out[o0 + Cout] = (_Float16)p1;
}

// ------------------------------------------------------------------
// att[b,e] = sigmoid(featsT[b,e,:] . aw[e,:] + ab[e])
// ------------------------------------------------------------------
__global__ void attention_kernel(const _Float16* __restrict__ feats, // [B,E,F]
                                 const float* __restrict__ aw,       // [E,F]
                                 const float* __restrict__ ab,       // [E]
                                 float* __restrict__ att, int Bn, int E, int F) {
  int i = blockIdx.x * blockDim.x + threadIdx.x;
  if (i >= Bn * E) return;
  int b = i / E, e = i - b * E;
  float s = ab[e];
  const _Float16* f = feats + ((size_t)b * E + e) * F;
  const float* w = aw + (size_t)e * F;
  for (int j = 0; j < F; ++j) s += (float)f[j] * w[j];
  att[i] = 1.f / (1.f + __expf(-s));
}

// ------------------------------------------------------------------
// A-fragment loader: row-major-along-K row, optional row scale.
// ------------------------------------------------------------------
DEV v16h load_a_row(const _Float16* row, int kb, int K, int grp, float scale) {
  v16h a;
  const int b0 = kb + (grp << 3), b1 = kb + 16 + (grp << 3);
  if (kb + 32 <= K) {
    v8h_u lo = *(const v8h_u*)(row + b0);
    v8h_u hi = *(const v8h_u*)(row + b1);
#pragma unroll
    for (int i = 0; i < 8; ++i) {
      a[i]     = (_Float16)((float)lo[i] * scale);
      a[8 + i] = (_Float16)((float)hi[i] * scale);
    }
  } else {
#pragma unroll
    for (int i = 0; i < 8; ++i) {
      a[i]     = (b0 + i < K) ? (_Float16)((float)row[b0 + i] * scale) : (_Float16)0.f;
      a[8 + i] = (b1 + i < K) ? (_Float16)((float)row[b1 + i] * scale) : (_Float16)0.f;
    }
  }
  return a;
}

// ==================================================================
// Batched GEMM: out[m,e,:] = act((A[m,e,:]*att[m,e]) @ BT[e]^T + bias[e])
// A: f16 [M=32,E,K]; BT: f16 [E,N,K] (pre-transposed); out f16/f32.
// Block = 128 thr (4 waves) computes 32 x 64 tile for one expert.
// B tile (64 rows x 32 K) staged in LDS by the Tensor Data Mover,
// waited with s_wait_tensorcnt, fragments read via ds_load_b128.
// grid: x = ceil(N/64), z = E.
// ==================================================================
__global__ void gemm_wmma_kernel(const _Float16* __restrict__ A,
                                 const _Float16* __restrict__ BT,
                                 const float* __restrict__ bias,
                                 const float* __restrict__ att,   // [M,E] or null
                                 void* __restrict__ out,
                                 int E, int K, int N, int relu, int out_f32) {
  __shared__ _Float16 shB[64 * 32];
  const int tid = threadIdx.x;
  const int wv = tid >> 5, lane = tid & 31;
  const int l15 = lane & 15, grp = lane >> 4;
  const int n0base = blockIdx.x << 6;
  const int e = blockIdx.z;
  const int n = n0base + (wv << 4) + l15;                 // this wave's column
  const float sc0 = att ? att[(size_t)l15 * E + e] : 1.0f;
  const float sc1 = att ? att[(size_t)(16 + l15) * E + e] : 1.0f;
  const _Float16* A0 = A + ((size_t)l15 * E + e) * K;          // rows 0..15
  const _Float16* A1 = A + ((size_t)(16 + l15) * E + e) * K;   // rows 16..31
  const _Float16* Btile = BT + ((size_t)e * N + n0base) * K;
  // LDS byte offset of shB: generic LDS pointers carry the offset in addr[31:0]
  const unsigned lds_off = (unsigned)(uintptr_t)(&shB[0]);

  v8f acc0 = {}, acc1 = {};
  for (int kb = 0; kb < K; kb += 32) {
    // ---- TDM: stage 64(N) x 32(K) f16 B tile into LDS (wave 0 issues) ----
    if (wv == 0) {
      tdm_load_2d_f16(lds_off, (const void*)(Btile + kb),
                      /*tensor_d0=*/(unsigned)(K - kb),
                      /*tensor_d1=*/(unsigned)(N - n0base),
                      /*tile_d0=*/32u, /*tile_d1=*/64u,
                      /*stride0=*/(unsigned long long)K);
      __builtin_amdgcn_s_wait_tensorcnt(0);
    }
    __syncthreads();
    // The TDM writes shB behind the compiler's back (descriptor carries only
    // an integer LDS offset). Escape the allocation + memory clobber so the
    // ds_load fragments below cannot be folded to undef / deleted.
    {
      _Float16* p = &shB[0];
      asm volatile("" : "+v"(p) : : "memory");
    }
    // ---- fragments ----
    v16h a0 = load_a_row(A0, kb, K, grp, sc0);
    v16h a1 = load_a_row(A1, kb, K, grp, sc1);
    v16h bf;
    {
      const _Float16* br = &shB[((wv << 4) + l15) * 32 + (grp << 3)];
      v8h lo = *(const v8h*)(br);
      v8h hi = *(const v8h*)(br + 16);
#pragma unroll
      for (int i = 0; i < 8; ++i) { bf[i] = lo[i]; bf[8 + i] = hi[i]; }
    }
    acc0 = __builtin_amdgcn_wmma_f32_16x16x32_f16(false, a0, false, bf,
                                                  (short)0, acc0, false, false);
    acc1 = __builtin_amdgcn_wmma_f32_16x16x32_f16(false, a1, false, bf,
                                                  (short)0, acc1, false, false);
    __syncthreads();                                      // before LDS reuse
  }
  if (n < N) {
    float bv = bias[(size_t)e * N + n];
#pragma unroll
    for (int r = 0; r < 8; ++r) {
      int m0 = r + (grp << 3);                            // C/D layout row, tile 0
      float v0 = acc0[r] + bv;
      float v1 = acc1[r] + bv;
      if (relu) { v0 = fmaxf(v0, 0.f); v1 = fmaxf(v1, 0.f); }
      size_t oi0 = ((size_t)m0 * E + e) * N + n;
      size_t oi1 = ((size_t)(16 + m0) * E + e) * N + n;
      if (out_f32) {
        ((float*)out)[oi0] = v0;
        ((float*)out)[oi1] = v1;
      } else {
        ((_Float16*)out)[oi0] = (_Float16)v0;
        ((_Float16*)out)[oi1] = (_Float16)v1;
      }
    }
  }
}

// ==================================================================
// Host-side orchestration
// ==================================================================
extern "C" void kernel_launch(void* const* d_in, const int* in_sizes, int n_in,
                              void* d_out, int out_size, void* d_ws, size_t ws_size,
                              hipStream_t stream) {
  (void)in_sizes; (void)n_in; (void)out_size; (void)ws_size;
  const int B = 32, E = 128, F = 196;
  const int C[5] = {3, 32, 64, 128, 128};
  const int HH[4] = {224, 112, 56, 28};
  const int EXPD[6] = {196, 196, 196, 98, 24, 16};
  const int FIND[4] = {2048, 2038, 2028, 53};

  const float* x = (const float*)d_in[0];
  const float* cw[4] = {(const float*)d_in[1], (const float*)d_in[3],
                        (const float*)d_in[5], (const float*)d_in[7]};
  const float* cb[4] = {(const float*)d_in[2], (const float*)d_in[4],
                        (const float*)d_in[6], (const float*)d_in[8]};
  const float* aw = (const float*)d_in[9];
  const float* ab = (const float*)d_in[10];
  const float* ew[5] = {(const float*)d_in[11], (const float*)d_in[13],
                        (const float*)d_in[15], (const float*)d_in[17],
                        (const float*)d_in[19]};
  const float* eb[5] = {(const float*)d_in[12], (const float*)d_in[14],
                        (const float*)d_in[16], (const float*)d_in[18],
                        (const float*)d_in[20]};
  const float* fw[3] = {(const float*)d_in[21], (const float*)d_in[23],
                        (const float*)d_in[25]};
  const float* fb[3] = {(const float*)d_in[22], (const float*)d_in[24],
                        (const float*)d_in[26]};

  char* base = (char*)d_ws;
  size_t off = 0;
  auto alloc = [&](size_t bytes) -> void* {
    void* p = base + off;
    off = (off + bytes + 255) & ~(size_t)255;
    return p;
  };
  _Float16* x16 = (_Float16*)alloc((size_t)B * 3 * 224 * 224 * 2);       // NHWC
  _Float16* cw16[4];
  for (int i = 0; i < 4; ++i)
    cw16[i] = (_Float16*)alloc((size_t)C[i + 1] * 9 * C[i] * 2);         // [Cout,9,Cin]
  _Float16* pool[4];                                                      // NHWC pooled
  for (int i = 0; i < 4; ++i)
    pool[i] = (_Float16*)alloc((size_t)B * (HH[i] / 2) * (HH[i] / 2) * C[i + 1] * 2);
  _Float16* featsT = (_Float16*)alloc((size_t)B * E * F * 2);            // [B,E,F]
  float* attb = (float*)alloc((size_t)B * E * 4);
  _Float16* ewT[5];
  for (int i = 0; i < 5; ++i)
    ewT[i] = (_Float16*)alloc((size_t)E * EXPD[i + 1] * EXPD[i] * 2);    // [E,N,K]
  _Float16* hbuf[5];
  for (int i = 0; i < 5; ++i)
    hbuf[i] = (_Float16*)alloc((size_t)B * E * EXPD[i + 1] * 2);
  _Float16* fwT[3];
  for (int i = 0; i < 3; ++i)
    fwT[i] = (_Float16*)alloc((size_t)FIND[i + 1] * FIND[i] * 2);        // [N,K]
  _Float16* gbuf[2];
  gbuf[0] = (_Float16*)alloc((size_t)B * FIND[1] * 2);
  gbuf[1] = (_Float16*)alloc((size_t)B * FIND[2] * 2);

  auto blocks = [](size_t n) { return dim3((unsigned)((n + 255) / 256)); };

  // ---- layout conversions (f32 -> f16, K-major) ----
  {
    size_t n = (size_t)B * 3 * 224 * 224;
    cvt_x_nhwc_kernel<<<blocks(n), 256, 0, stream>>>(x, x16, 3, 224, 224, (int)n);
  }
  for (int i = 0; i < 4; ++i) {
    size_t n = (size_t)C[i + 1] * 9 * C[i];
    cvt_convw_kernel<<<blocks(n), 256, 0, stream>>>(cw[i], cw16[i], C[i], (int)n);
  }
  for (int i = 0; i < 5; ++i) {
    size_t n = (size_t)E * EXPD[i] * EXPD[i + 1];
    cvt_wT_kernel<<<blocks(n), 256, 0, stream>>>(ew[i], ewT[i], EXPD[i], EXPD[i + 1], (int)n);
  }
  for (int i = 0; i < 3; ++i) {
    size_t n = (size_t)FIND[i] * FIND[i + 1];
    cvt_wT_kernel<<<blocks(n), 256, 0, stream>>>(fw[i], fwT[i], FIND[i], FIND[i + 1], (int)n);
  }

  // ---- CNN block: 4x fused conv+relu+pool (WMMA implicit GEMM, NHWC) ----
  const _Float16* cin_p = x16;
  for (int i = 0; i < 4; ++i) {
    int H = HH[i];
    dim3 grid((H / 4) * (H / 4), C[i + 1] / 16, B);
    conv_wmma_kernel<<<grid, 32, 0, stream>>>(cin_p, cw16[i], cb[i], pool[i],
                                              C[i], H, H, C[i + 1]);
    cin_p = pool[i];
  }

  // ---- feats: [B,14,14,128] (NHWC) -> [B,E,F] ----
  {
    size_t n = (size_t)B * E * F;
    feats_transpose_kernel<<<blocks(n), 256, 0, stream>>>(pool[3], featsT, E, F, (int)n);
  }

  // ---- attention ----
  attention_kernel<<<blocks((size_t)B * E), 256, 0, stream>>>(featsT, aw, ab, attb, B, E, F);

  // ---- 128 stacked expert MLPs (batched WMMA GEMM + TDM-staged B) ----
  const _Float16* ain = featsT;
  for (int i = 0; i < 5; ++i) {
    int K = EXPD[i], N = EXPD[i + 1];
    dim3 grid((N + 63) / 64, 1, E);
    gemm_wmma_kernel<<<grid, 128, 0, stream>>>(
        ain, ewT[i], eb[i], (i == 0) ? attb : nullptr, hbuf[i],
        E, K, N, /*relu=*/1, /*out_f32=*/0);
    ain = hbuf[i];
  }

  // ---- final dense block: 2048 -> 2038 -> 2028 -> 53 ----
  const _Float16* gin = hbuf[4];                         // [B,E,16] == [B,2048]
  for (int i = 0; i < 3; ++i) {
    int K = FIND[i], N = FIND[i + 1];
    int last = (i == 2);
    dim3 grid((N + 63) / 64, 1, 1);
    gemm_wmma_kernel<<<grid, 128, 0, stream>>>(
        gin, fwT[i], fb[i], nullptr, last ? d_out : (void*)gbuf[i],
        1, K, N, /*relu=*/!last, /*out_f32=*/last);
    if (!last) gin = gbuf[i];
  }
}
